// DP3Encoder_17051020165780
// MI455X (gfx1250) — matrix operand order repaired and checked by reading, not compiled, not run on hardware
//
#include <hip/hip_runtime.h>

typedef __attribute__((ext_vector_type(16))) _Float16 v16h;
typedef __attribute__((ext_vector_type(8)))  _Float16 v8h;
typedef __attribute__((ext_vector_type(8)))  float    v8f;

#define KS 256
#define FPS_THREADS 1024

// ---------------------------------------------------------------------------
// Weight prep: W[K,N] (f32, row-major) -> Wt[N][Kpad] (f16, transposed, padded)
// ---------------------------------------------------------------------------
__global__ void prep_wt(const float* __restrict__ W, _Float16* __restrict__ Wt,
                        int K, int N, int Kpad) {
  int i = blockIdx.x * blockDim.x + threadIdx.x;
  int total = N * Kpad;
  if (i >= total) return;
  int n = i / Kpad, k = i - n * Kpad;
  float v = (k < K) ? W[k * N + n] : 0.0f;
  Wt[i] = (_Float16)v;
}

// ---------------------------------------------------------------------------
// Farthest point sampling: one 1024-thread block per batch.
// min_d lives in LDS (256 KB of the 320 KB WGP LDS); points stream from L2.
// Each thread owns 4 consecutive points: 3x global_load_b128 for xyz,
// 1x ds_load_b128/ds_store_b128 for min_d.
// ---------------------------------------------------------------------------
__global__ __launch_bounds__(FPS_THREADS)
void fps_kernel(const float* __restrict__ pc, int* __restrict__ idx,
                int N, int Ksamp) {
  extern __shared__ char smem[];
  float* mind = (float*)smem;                         // N floats
  float* rv   = (float*)(smem + (size_t)N * 4);       // 32
  int*   ri   = (int*)(rv + 32);                      // 32
  float* bc   = (float*)(ri + 32);                    // 3 (broadcast point)

  const int tid  = threadIdx.x;
  const int lane = tid & 31;
  const int wid  = tid >> 5;
  const float* p = pc + (size_t)blockIdx.x * N * 3;

  if (tid == 0) {
    idx[blockIdx.x * Ksamp] = 0;
    bc[0] = p[0]; bc[1] = p[1]; bc[2] = p[2];
  }
  for (int i = tid * 4; i < N; i += FPS_THREADS * 4) {
    float4 init = {1.0e10f, 1.0e10f, 1.0e10f, 1.0e10f};
    *(float4*)(mind + i) = init;
  }
  __syncthreads();

  for (int s = 1; s < Ksamp; ++s) {
    const float px = bc[0], py = bc[1], pz = bc[2];
    float bestv = -1.0f; int besti = 0;
    for (int i0 = tid * 4; i0 < N; i0 += FPS_THREADS * 4) {
      const float4* q = (const float4*)(p + (size_t)3 * i0);
      const float4 q0 = q[0], q1 = q[1], q2 = q[2];
      float4 m4 = *(float4*)(mind + i0);
      float dx, dy, dz;
      dx = q0.x - px; dy = q0.y - py; dz = q0.z - pz;
      m4.x = fminf(m4.x, dx*dx + dy*dy + dz*dz);
      dx = q0.w - px; dy = q1.x - py; dz = q1.y - pz;
      m4.y = fminf(m4.y, dx*dx + dy*dy + dz*dz);
      dx = q1.z - px; dy = q1.w - py; dz = q2.x - pz;
      m4.z = fminf(m4.z, dx*dx + dy*dy + dz*dz);
      dx = q2.y - px; dy = q2.z - py; dz = q2.w - pz;
      m4.w = fminf(m4.w, dx*dx + dy*dy + dz*dz);
      *(float4*)(mind + i0) = m4;
      // strict > keeps the earliest index (jnp.argmax tie rule)
      if (m4.x > bestv) { bestv = m4.x; besti = i0; }
      if (m4.y > bestv) { bestv = m4.y; besti = i0 + 1; }
      if (m4.z > bestv) { bestv = m4.z; besti = i0 + 2; }
      if (m4.w > bestv) { bestv = m4.w; besti = i0 + 3; }
    }
    // wave32 argmax reduction (ties -> smaller index)
    #pragma unroll
    for (int off = 16; off > 0; off >>= 1) {
      float ov = __shfl_xor(bestv, off, 32);
      int   oi = __shfl_xor(besti, off, 32);
      if (ov > bestv || (ov == bestv && oi < besti)) { bestv = ov; besti = oi; }
    }
    if (lane == 0) { rv[wid] = bestv; ri[wid] = besti; }
    __syncthreads();
    if (wid == 0) {
      float v = rv[lane]; int ii = ri[lane];
      #pragma unroll
      for (int off = 16; off > 0; off >>= 1) {
        float ov = __shfl_xor(v, off, 32);
        int   oi = __shfl_xor(ii, off, 32);
        if (ov > v || (ov == v && oi < ii)) { v = ov; ii = oi; }
      }
      if (lane == 0) {
        idx[blockIdx.x * Ksamp + s] = ii;
        bc[0] = p[3*ii+0]; bc[1] = p[3*ii+1]; bc[2] = p[3*ii+2];
      }
    }
    __syncthreads();
  }
}

// ---------------------------------------------------------------------------
// WMMA fragment loads, all through ext-vector types so extraction stays in
// registers (no scalar ds_load_u16 fallback).
// A: LDS row-major f16 [256][Kin]. lane<16 holds row M0+lane, K 0..15;
//    lane>=16 holds K 16..31  -> 2x ds_load_b128 + register concat.
// B: global f16 Wt[N][Kin] (transposed): lane<16 = col N0+lane K 0..15,
//    lane>=16 K 16..31 -> one v16h load (2x global_load_b128).
// C: v8f, element r -> row M0 + r + (lane>=16 ? 8 : 0), col N0 + (lane&15).
// ---------------------------------------------------------------------------
__device__ __forceinline__ v16h load_a_frag(const _Float16* arow, int k0) {
  const v8h a0 = *(const v8h*)(arow + k0);
  const v8h a1 = *(const v8h*)(arow + k0 + 8);
  return __builtin_shufflevector(a0, a1, 0, 1, 2, 3, 4, 5, 6, 7,
                                         8, 9, 10, 11, 12, 13, 14, 15);
}

template <int Kin, int Nout>
__device__ __forceinline__ void gemm_tile_store(
    const _Float16* __restrict__ in,
    const _Float16* __restrict__ Wt,
    const float* __restrict__ bias,
    _Float16* __restrict__ out, int wave, int lane) {
  constexpr int Ntiles = Nout >> 4;
  constexpr int total  = Ntiles << 4;          // Ntiles * 16 M-tiles
  const int l15 = lane & 15;
  const int hi  = lane >> 4;
  for (int pcs = wave; pcs < total; pcs += 8) {
    const int mt = pcs & 15;
    const int nt = pcs >> 4;
    const int M0 = mt << 4, N0 = nt << 4;
    const _Float16* arow = in + (M0 + l15) * Kin + (hi ? 16 : 0);
    const int col = N0 + l15;
    const _Float16* brow = Wt + col * Kin + (hi ? 16 : 0);
    __builtin_prefetch(brow, 0, 1);
    v8f c = {};
    #pragma unroll
    for (int k0 = 0; k0 < Kin; k0 += 32) {
      const v16h a = load_a_frag(arow, k0);
      const v16h b = *(const v16h*)(brow + k0);
      c = __builtin_amdgcn_wmma_f32_16x16x32_f16(false, a, false, b,
                                                 (short)0, c, false, false);
    }
    const float bv = bias[col];
    const int rbase = M0 + (hi ? 8 : 0);
    #pragma unroll
    for (int r = 0; r < 8; ++r)
      out[(rbase + r) * Nout + col] = (_Float16)(c[r] + bv);
  }
}

// Final layer: GEMM fused with column-max pooling over the 256 rows.
// Waves own exclusive N-tiles; B fragments hoisted to registers across M-tiles.
__device__ __forceinline__ void gemm_colmax(
    const _Float16* __restrict__ in,              // [256][256]
    const _Float16* __restrict__ Wt,              // [512][256]
    const float* __restrict__ bias,
    float* __restrict__ feat, int wave, int lane) {
  constexpr int Kin = 256;
  const int l15 = lane & 15;
  const int hi  = lane >> 4;
  for (int nt = wave; nt < 32; nt += 8) {
    const int col = (nt << 4) + l15;
    const _Float16* brow = Wt + col * Kin + (hi ? 16 : 0);
    v16h bfrag[8];
    #pragma unroll
    for (int kb = 0; kb < 8; ++kb)
      bfrag[kb] = *(const v16h*)(brow + kb * 32);
    float cmax = -3.0e38f;
    for (int mt = 0; mt < 16; ++mt) {
      const _Float16* arow = in + ((mt << 4) + l15) * Kin + (hi ? 16 : 0);
      v8f c = {};
      #pragma unroll
      for (int kb = 0; kb < 8; ++kb) {
        const v16h a = load_a_frag(arow, kb * 32);
        c = __builtin_amdgcn_wmma_f32_16x16x32_f16(false, a, false, bfrag[kb],
                                                   (short)0, c, false, false);
      }
      #pragma unroll
      for (int r = 0; r < 8; ++r) cmax = fmaxf(cmax, c[r]);
    }
    cmax += bias[col];
    const float o = __shfl_xor(cmax, 16, 32);     // partner lane shares `col`
    cmax = fmaxf(cmax, o);
    if (hi == 0) feat[col] = cmax;
  }
}

// Per-row LayerNorm + ReLU on an LDS f16 buffer (one thread per row).
// v8h loads -> ds_load_b128; rolled outer loop keeps code size small.
template <int NC>
__device__ __forceinline__ void ln_relu_rows(
    _Float16* buf, const float* __restrict__ g,
    const float* __restrict__ be, int tid) {
  _Float16* rowp = buf + tid * NC;
  float s = 0.f, s2 = 0.f;
  for (int i = 0; i < NC; i += 8) {
    const v8h u = *(const v8h*)(rowp + i);
    #pragma unroll
    for (int j = 0; j < 8; ++j) { const float v = (float)u[j]; s += v; s2 += v*v; }
  }
  constexpr float inv = 1.0f / (float)NC;
  const float m = s * inv;
  float var = s2 * inv - m * m;
  var = fmaxf(var, 0.0f);
  const float r = rsqrtf(var + 1e-5f);
  for (int i = 0; i < NC; i += 8) {
    const v8h u = *(const v8h*)(rowp + i);
    const float4 g0 = *(const float4*)(g + i);
    const float4 g1 = *(const float4*)(g + i + 4);
    const float4 e0 = *(const float4*)(be + i);
    const float4 e1 = *(const float4*)(be + i + 4);
    v8h w;
    w[0] = (_Float16)fmaxf(((float)u[0] - m) * r * g0.x + e0.x, 0.0f);
    w[1] = (_Float16)fmaxf(((float)u[1] - m) * r * g0.y + e0.y, 0.0f);
    w[2] = (_Float16)fmaxf(((float)u[2] - m) * r * g0.z + e0.z, 0.0f);
    w[3] = (_Float16)fmaxf(((float)u[3] - m) * r * g0.w + e0.w, 0.0f);
    w[4] = (_Float16)fmaxf(((float)u[4] - m) * r * g1.x + e1.x, 0.0f);
    w[5] = (_Float16)fmaxf(((float)u[5] - m) * r * g1.y + e1.y, 0.0f);
    w[6] = (_Float16)fmaxf(((float)u[6] - m) * r * g1.z + e1.z, 0.0f);
    w[7] = (_Float16)fmaxf(((float)u[7] - m) * r * g1.w + e1.w, 0.0f);
    *(v8h*)(rowp + i) = w;
  }
}

// ---------------------------------------------------------------------------
// Encoder: one 256-thread block (8 waves) per batch. Gather -> 4 WMMA layers
// (LN+ReLU between) -> max-pool -> pool FC + LN -> state MLP -> d_out[b,320].
// ---------------------------------------------------------------------------
__global__ __launch_bounds__(256)
void encoder_kernel(
    const float* __restrict__ pc, const int* __restrict__ idx,
    const _Float16* __restrict__ W0t, const float* __restrict__ b0,
    const float* __restrict__ g0, const float* __restrict__ be0,
    const _Float16* __restrict__ W1t, const float* __restrict__ b1,
    const float* __restrict__ g1, const float* __restrict__ be1,
    const _Float16* __restrict__ W2t, const float* __restrict__ b2,
    const float* __restrict__ g2, const float* __restrict__ be2,
    const _Float16* __restrict__ W3t, const float* __restrict__ b3,
    const _Float16* __restrict__ Wpt, const float* __restrict__ bp,
    const float* __restrict__ gp, const float* __restrict__ bep,
    const float* __restrict__ sp, const float* __restrict__ sq,
    const float* __restrict__ sg,
    const float* __restrict__ Ws0, const float* __restrict__ bs0,
    const float* __restrict__ Ws1, const float* __restrict__ bs1,
    float* __restrict__ out, int N) {
  extern __shared__ char smem[];
  _Float16* actA  = (_Float16*)smem;                         // 256*128 f16
  _Float16* actB  = (_Float16*)(smem + 256*128*2);           // 256*256 f16
  float*    feat  = (float*)(smem + 256*128*2 + 256*256*2);  // 512
  float*    pooled= feat + 512;                              // 256
  float*    shid  = pooled + 256;                            // 64

  const int b    = blockIdx.x;
  const int tid  = threadIdx.x;
  const int wave = tid >> 5;
  const int lane = tid & 31;

  // Gather sampled points, pad K 3->32 with zeros (W0t rows padded to match).
  {
    const int j = idx[b * KS + tid];
    const size_t pbase = (size_t)b * N * 3 + (size_t)j * 3;
    _Float16* rp = actA + tid * 32;
    v8h z = {};
    v8h first = z;
    first[0] = (_Float16)pc[pbase + 0];
    first[1] = (_Float16)pc[pbase + 1];
    first[2] = (_Float16)pc[pbase + 2];
    *(v8h*)(rp +  0) = first;
    *(v8h*)(rp +  8) = z;
    *(v8h*)(rp + 16) = z;
    *(v8h*)(rp + 24) = z;
  }
  __syncthreads();

  gemm_tile_store< 32,  64>(actA, W0t, b0, actB, wave, lane); __syncthreads();
  ln_relu_rows< 64>(actB, g0, be0, tid);                      __syncthreads();
  gemm_tile_store< 64, 128>(actB, W1t, b1, actA, wave, lane); __syncthreads();
  ln_relu_rows<128>(actA, g1, be1, tid);                      __syncthreads();
  gemm_tile_store<128, 256>(actA, W2t, b2, actB, wave, lane); __syncthreads();
  ln_relu_rows<256>(actB, g2, be2, tid);                      __syncthreads();
  gemm_colmax(actB, W3t, b3, feat, wave, lane);               __syncthreads();

  // pooled = feat @ Wp + bp  (one thread per output column; feat broadcast)
  {
    const _Float16* wcol = Wpt + tid * 512;
    float acc = bp[tid];
    for (int k = 0; k < 512; k += 8) {
      const v8h u = *(const v8h*)(wcol + k);
      const float4 fa = *(const float4*)(feat + k);
      const float4 fb = *(const float4*)(feat + k + 4);
      acc += fa.x * (float)u[0] + fa.y * (float)u[1] +
             fa.z * (float)u[2] + fa.w * (float)u[3] +
             fb.x * (float)u[4] + fb.y * (float)u[5] +
             fb.z * (float)u[6] + fb.w * (float)u[7];
    }
    pooled[tid] = acc;
  }
  __syncthreads();

  // LayerNorm over pooled[256], write first 256 outputs.
  {
    float s = 0.f, s2 = 0.f;
    for (int c = 0; c < 256; c += 4) {
      const float4 v4 = *(const float4*)(pooled + c);
      s  += v4.x + v4.y + v4.z + v4.w;
      s2 += v4.x*v4.x + v4.y*v4.y + v4.z*v4.z + v4.w*v4.w;
    }
    const float m = s * (1.0f/256.0f);
    float var = s2 * (1.0f/256.0f) - m * m;
    var = fmaxf(var, 0.0f);
    const float r = rsqrtf(var + 1e-5f);
    out[b * 320 + tid] = (pooled[tid] - m) * r * gp[tid] + bep[tid];
  }

  // State MLP: 9 -> 64 (relu) -> 64, outputs 256..319.
  if (tid < 64) {
    float st[9];
    st[0]=sp[b*3+0]; st[1]=sp[b*3+1]; st[2]=sp[b*3+2];
    st[3]=sq[b*4+0]; st[4]=sq[b*4+1]; st[5]=sq[b*4+2]; st[6]=sq[b*4+3];
    st[7]=sg[b*2+0]; st[8]=sg[b*2+1];
    float acc = bs0[tid];
    #pragma unroll
    for (int k = 0; k < 9; ++k) acc += st[k] * Ws0[k*64 + tid];
    shid[tid] = fmaxf(acc, 0.0f);
  }
  __syncthreads();
  if (tid < 64) {
    float acc = bs1[tid];
    for (int k = 0; k < 64; ++k) acc += shid[k] * Ws1[k*64 + tid];
    out[b * 320 + 256 + tid] = acc;
  }
}

// ---------------------------------------------------------------------------
extern "C" void kernel_launch(void* const* d_in, const int* in_sizes, int n_in,
                              void* d_out, int out_size, void* d_ws, size_t ws_size,
                              hipStream_t stream) {
  (void)n_in; (void)out_size; (void)ws_size;
  const float* pc  = (const float*)d_in[0];
  const float* sp  = (const float*)d_in[1];
  const float* sq  = (const float*)d_in[2];
  const float* sg  = (const float*)d_in[3];
  const float* W0  = (const float*)d_in[4];  const float* b0  = (const float*)d_in[5];
  const float* g0  = (const float*)d_in[6];  const float* be0 = (const float*)d_in[7];
  const float* W1  = (const float*)d_in[8];  const float* b1  = (const float*)d_in[9];
  const float* g1  = (const float*)d_in[10]; const float* be1 = (const float*)d_in[11];
  const float* W2  = (const float*)d_in[12]; const float* b2  = (const float*)d_in[13];
  const float* g2  = (const float*)d_in[14]; const float* be2 = (const float*)d_in[15];
  const float* W3  = (const float*)d_in[16]; const float* b3  = (const float*)d_in[17];
  const float* Wp  = (const float*)d_in[18]; const float* bp  = (const float*)d_in[19];
  const float* gp  = (const float*)d_in[20]; const float* bep = (const float*)d_in[21];
  const float* Ws0 = (const float*)d_in[22]; const float* bs0 = (const float*)d_in[23];
  const float* Ws1 = (const float*)d_in[24]; const float* bs1 = (const float*)d_in[25];

  const int B = in_sizes[1] / 3;               // 128
  const int N = in_sizes[0] / (3 * B);         // 65536

  // Workspace: FPS indices + f16 transposed weights (< 1 MB total).
  char* ws = (char*)d_ws;
  int* idx = (int*)ws;
  size_t off = (size_t)B * KS * sizeof(int);
  _Float16* W0t = (_Float16*)(ws + off); off += (size_t)64  * 32  * 2;
  _Float16* W1t = (_Float16*)(ws + off); off += (size_t)128 * 64  * 2;
  _Float16* W2t = (_Float16*)(ws + off); off += (size_t)256 * 128 * 2;
  _Float16* W3t = (_Float16*)(ws + off); off += (size_t)512 * 256 * 2;
  _Float16* Wpt = (_Float16*)(ws + off); off += (size_t)256 * 512 * 2;

  auto prep = [&](const float* W, _Float16* Wt, int K, int Nn, int Kpad) {
    const int total = Nn * Kpad;
    prep_wt<<<(total + 255) / 256, 256, 0, stream>>>(W, Wt, K, Nn, Kpad);
  };
  prep(W0, W0t,   3,  64,  32);
  prep(W1, W1t,  64, 128,  64);
  prep(W2, W2t, 128, 256, 128);
  prep(W3, W3t, 256, 512, 256);
  prep(Wp, Wpt, 512, 256, 512);

  const size_t fps_lds = (size_t)N * 4 + 32 * 4 + 32 * 4 + 4 * 4;
  fps_kernel<<<B, FPS_THREADS, fps_lds, stream>>>(pc, idx, N, KS);

  const size_t enc_lds = 256*128*2 + 256*256*2 + 512*4 + 256*4 + 64*4;
  encoder_kernel<<<B, 256, enc_lds, stream>>>(
      pc, idx, W0t, b0, g0, be0, W1t, b1, g1, be1, W2t, b2, g2, be2,
      W3t, b3, Wpt, bp, gp, bep, sp, sq, sg, Ws0, bs0, Ws1, bs1,
      (float*)d_out, N);
}
